// BertWordPair_10582799417749
// MI455X (gfx1250) — matrix operand order repaired and checked by reading, not compiled, not run on hardware
//
#include <hip/hip_runtime.h>
#include <hip/hip_bf16.h>

typedef _Float16 half2v __attribute__((ext_vector_type(2)));
typedef _Float16 half8  __attribute__((ext_vector_type(8)));
typedef _Float16 v16h   __attribute__((ext_vector_type(16)));
typedef float    v8f    __attribute__((ext_vector_type(8)));

constexpr int B_ = 8, S_ = 768, H_ = 4, D_ = 256;
constexpr int HALF_ = D_ / 2;          // 128 rotation pairs per row
constexpr int LDA_ = D_ + 8;           // LDS row pitch in halves (16B pad)

// ---------------------------------------------------------------------------
// Phase 1: interleaved RoPE rotate (+pos and -pos), f32 -> f16, and transpose
// layout (b,s,h,d) -> (b,h,s,d) so each (b,h) GEMM slice has contiguous rows.
// out[2i]   = x[2i]*cos_i - x[2i+1]*sin_i
// out[2i+1] = x[2i+1]*cos_i + x[2i]*sin_i        (negative pos: sin -> -sin)
// ---------------------------------------------------------------------------
__global__ __launch_bounds__(128) void rope_kernel(
    const float* __restrict__ qw, const float* __restrict__ kw,
    const int* __restrict__ tok,
    _Float16* __restrict__ qp, _Float16* __restrict__ qn,
    _Float16* __restrict__ kp, _Float16* __restrict__ kn)
{
    const int row = blockIdx.x;                 // b*S*H + s*H + h
    const int h = row % H_;
    const int s = (row / H_) % S_;
    const int b = row / (H_ * S_);
    const int i = threadIdx.x;                  // pair index 0..127

    const float pos = (float)tok[s];
    // inv_freq_i = 10000^(-2i/D) = exp(-i * ln(10000)/128)
    const float inv = __expf(-(float)i * (9.210340371976184f / (float)HALF_));
    const float theta = pos * inv;
    float sn, cs;
    __sincosf(theta, &sn, &cs);

    const size_t in_base  = (((size_t)(b * S_ + s) * H_ + h) * D_) + 2 * i;
    const size_t out_base = (((size_t)(b * H_ + h) * S_ + s) * D_) + 2 * i;

    const float2 x = *(const float2*)(qw + in_base);
    const float2 y = *(const float2*)(kw + in_base);

    half2v v;
    v.x = (_Float16)(x.x * cs - x.y * sn);  v.y = (_Float16)(x.y * cs + x.x * sn);
    *(half2v*)(qp + out_base) = v;
    v.x = (_Float16)(x.x * cs + x.y * sn);  v.y = (_Float16)(x.y * cs - x.x * sn);
    *(half2v*)(qn + out_base) = v;
    v.x = (_Float16)(y.x * cs - y.y * sn);  v.y = (_Float16)(y.y * cs + y.x * sn);
    *(half2v*)(kp + out_base) = v;
    v.x = (_Float16)(y.x * cs + y.y * sn);  v.y = (_Float16)(y.y * cs - y.x * sn);
    *(half2v*)(kn + out_base) = v;
}

// ---------------------------------------------------------------------------
// Phase 2: per (b,h), tiled GEMM  out[m,n] = dot(A[m,:], B[n,:]) over D=256.
// Block tile = 32(m) x 64(n), 8 waves, one 16x16 WMMA tile per wave.
// thread_id is constant over aligned 128-runs -> mask is uniform per block,
// so we pick exactly one of {q_p.k_p, q_n.k_p, q_p.k_n} with scalar compares.
// ---------------------------------------------------------------------------
__global__ __launch_bounds__(256) void gemm_kernel(
    const _Float16* __restrict__ qp, const _Float16* __restrict__ qn,
    const _Float16* __restrict__ kp, const _Float16* __restrict__ kn,
    const int* __restrict__ tid_arr, float* __restrict__ out)
{
    __shared__ _Float16 As[32 * LDA_];
    __shared__ _Float16 Bs[64 * LDA_];

    const int nblk = blockIdx.x;          // 0..11  (64 cols each)
    const int mblk = blockIdx.y;          // 0..23  (32 rows each)
    const int bh   = blockIdx.z;          // b*H + h
    const int b = bh / H_, h = bh % H_;

    const int m0 = mblk * 32;
    const int n0 = nblk * 64;

    const int ti_r = tid_arr[m0];
    const int ti_c = tid_arr[n0];

    const _Float16* Asrc;
    const _Float16* Bsrc;
    if (ti_r > 0 && ti_r < ti_c)      { Asrc = qn; Bsrc = kp; }   // s_np
    else if (ti_c > 0 && ti_r > ti_c) { Asrc = qp; Bsrc = kn; }   // s_pn
    else                              { Asrc = qp; Bsrc = kp; }   // s_pp

    const size_t abase = ((size_t)bh * S_ + m0) * D_;
    const size_t bbase = ((size_t)bh * S_ + n0) * D_;

    const int t = threadIdx.x;            // 0..255

    // Stage A: 32 rows x 256 halves = 1024 x half8 chunks (4 per thread)
    #pragma unroll
    for (int it = 0; it < 4; ++it) {
        const int idx = t + it * 256;
        const int r = idx >> 5;           // 0..31
        const int c = idx & 31;           // 0..31 (8-half chunks)
        half8 v = *(const half8*)(Asrc + abase + (size_t)r * D_ + c * 8);
        *(half8*)(&As[r * LDA_ + c * 8]) = v;
    }
    // Stage B: 64 rows x 256 halves = 2048 chunks (8 per thread)
    #pragma unroll
    for (int it = 0; it < 8; ++it) {
        const int idx = t + it * 256;
        const int r = idx >> 5;           // 0..63
        const int c = idx & 31;
        half8 v = *(const half8*)(Bsrc + bbase + (size_t)r * D_ + c * 8);
        *(half8*)(&Bs[r * LDA_ + c * 8]) = v;
    }
    __syncthreads();

    const int wave = t >> 5;              // 0..7
    const int lane = t & 31;
    const int msub = wave >> 2;           // 0..1  (16-row sub-tile)
    const int nsub = wave & 3;            // 0..3  (16-col sub-tile)

    const int mn = lane & 15;             // A row / B col / C col index
    const int g  = lane >> 4;             // half-wave group

    const _Float16* arow = &As[(msub * 16 + mn) * LDA_];
    const _Float16* brow = &Bs[(nsub * 16 + mn) * LDA_];

    v8f c = {};
    #pragma unroll
    for (int k = 0; k < 8; ++k) {         // K = 8 x 32
        const int koff = k * 32;
        // A layout (ISA 16-bit A 16x32): lane holds K in
        //   [g*8, g*8+8)  and  [16+g*8, 16+g*8+8)
        half8 a0 = *(const half8*)(arow + koff + g * 8);
        half8 a1 = *(const half8*)(arow + koff + 16 + g * 8);
        // B layout (same K order as A): lane holds K in
        //   [g*8, g*8+8)  and  [16+g*8, 16+g*8+8)
        half8 b0 = *(const half8*)(brow + koff + g * 8);
        half8 b1 = *(const half8*)(brow + koff + 16 + g * 8);
        v16h a  = __builtin_shufflevector(a0, a1, 0,1,2,3,4,5,6,7,8,9,10,11,12,13,14,15);
        v16h bm = __builtin_shufflevector(b0, b1, 0,1,2,3,4,5,6,7,8,9,10,11,12,13,14,15);
        c = __builtin_amdgcn_wmma_f32_16x16x32_f16(
                /*neg_a=*/false, a, /*neg_b=*/false, bm,
                /*c_mod=*/(short)0, c, /*reuse_a=*/false, /*reuse_b=*/false);
    }

    // C layout: VGPR r -> M = r + 8*g ; N = lane & 15. out is (B, S, S, H).
    const int m = m0 + msub * 16 + 8 * g;
    const int n = n0 + nsub * 16 + mn;
    size_t obase = (((size_t)b * S_ + m) * S_ + n) * H_ + h;
    #pragma unroll
    for (int r = 0; r < 8; ++r) {
        __builtin_nontemporal_store(c[r], out + obase + (size_t)r * S_ * H_);
    }
}

extern "C" void kernel_launch(void* const* d_in, const int* in_sizes, int n_in,
                              void* d_out, int out_size, void* d_ws, size_t ws_size,
                              hipStream_t stream) {
    const float* qw  = (const float*)d_in[0];
    const float* kw  = (const float*)d_in[1];
    const int*   tok = (const int*)d_in[2];
    const int*   tid = (const int*)d_in[3];
    float* out = (float*)d_out;

    const size_t per = (size_t)B_ * H_ * S_ * D_;   // halves per rotated tensor
    _Float16* qp = (_Float16*)d_ws;
    _Float16* qn = qp + per;
    _Float16* kp = qn + per;
    _Float16* kn = kp + per;

    rope_kernel<<<B_ * S_ * H_, 128, 0, stream>>>(qw, kw, tok, qp, qn, kp, kn);

    dim3 grid(S_ / 64, S_ / 32, B_ * H_);
    gemm_kernel<<<grid, 256, 0, stream>>>(qp, qn, kp, kn, tid, out);
}
